// EGNNLayer_24610162606596
// MI455X (gfx1250) — matrix operand order repaired and checked by reading, not compiled, not run on hardware
//
#include <hip/hip_runtime.h>
#include <hip/hip_bf16.h>

#define N_NODES  50000
#define N_EDGES  800000
#define NODE_DIM 64
#define EDGE_DIM 16
#define HIDDEN   128
#define COORDS   2

#define WAVES_PER_BLOCK 4
#define BLOCK_THREADS   (WAVES_PER_BLOCK * 32)

// LDS tile strides (elements, even, spread across banks to avoid 4-way conflicts)
#define EDGE_ASTRIDE 164   // >= 160 (padded K for edge MLP input) and >= 128
#define NODE_ASTRIDE 196   // >= 192 (h | node_aggr)

typedef __attribute__((ext_vector_type(16))) __bf16 v16bf;
typedef __attribute__((ext_vector_type(8)))  float  v8f;

union AFragU { v16bf v; unsigned u[8]; };

__device__ __forceinline__ void lds_fence() {
  // per-wave producer->consumer ordering through LDS; DS ops are in-order in HW,
  // this just stops the compiler from reordering across the stage boundary.
  __builtin_amdgcn_wave_barrier();
}

// Load one 16x32 bf16 A-fragment from a row-major LDS tile.
// Layout (ISA 7.12.2): lanes 0-15 = rows M=0..15 holding K=kb+{0..7,16..23},
// lanes 16-31 = same rows holding K=kb+{8..15,24..31}; 2 bf16 per VGPR.
__device__ __forceinline__ v16bf load_a_frag(const __bf16* tile, int stride,
                                             int lane, int kbase) {
  AFragU f;
  const int m  = lane & 15;
  const int kb = kbase + ((lane >> 4) << 3);
  const unsigned* row = (const unsigned*)(tile + m * stride);
#pragma unroll
  for (int j = 0; j < 8; ++j) {
    const int koff = ((j & 3) << 1) + ((j >> 2) << 4);
    f.u[j] = row[(kb + koff) >> 1];
  }
  return f.v;
}

// C += A(16xK) * B(KxN) over KT k-tiles and NT n-tiles; B pre-swizzled so each
// lane loads 16 contiguous bf16 (32B) per fragment.
template<int KT, int NT>
__device__ __forceinline__ void wmma_gemm(const __bf16* in, int istride,
                                          const __bf16* W, int lane, v8f* acc) {
#pragma unroll
  for (int nt = 0; nt < NT; ++nt) {
    v8f z = {0.f, 0.f, 0.f, 0.f, 0.f, 0.f, 0.f, 0.f};
    acc[nt] = z;
  }
#pragma unroll
  for (int kt = 0; kt < KT; ++kt) {
    v16bf a = load_a_frag(in, istride, lane, kt * 32);
#pragma unroll
    for (int nt = 0; nt < NT; ++nt) {
      v16bf b = *(const v16bf*)(W + (((kt * NT + nt) * 32 + lane) << 4));
      acc[nt] = __builtin_amdgcn_wmma_f32_16x16x32_bf16(
          false, a, false, b, (short)0, acc[nt], false, false);
    }
  }
}

// Full layer: GEMM + bias (+ optional SiLU), writing bf16 activations to LDS.
// C layout: VGPR r -> M = r + 8*(lane>=16); N = nt*16 + (lane&15).
template<int KT, int NT, bool ACT>
__device__ __forceinline__ void mlp_layer(const __bf16* in, int istride,
                                          __bf16* out, int ostride,
                                          const __bf16* W, const float* bias,
                                          int lane) {
  v8f acc[NT];
  wmma_gemm<KT, NT>(in, istride, W, lane, acc);
  const int nlo = lane & 15;
  const int mb  = (lane >> 4) << 3;
#pragma unroll
  for (int nt = 0; nt < NT; ++nt) {
    const int n  = (nt << 4) + nlo;
    const float bv = bias[n];
#pragma unroll
    for (int r = 0; r < 8; ++r) {
      float v = acc[nt][r] + bv;
      if (ACT) v = v / (1.f + __expf(-v));   // SiLU
      out[(mb + r) * ostride + n] = (__bf16)v;
    }
  }
  lds_fence();
}

// ---------------------------------------------------------------------------
// Fused edge kernel: one wave = 16 edges.
// gather -> edge MLP -> attn -> scale -> node_aggr atomics -> coord MLP -> x atomics
// ---------------------------------------------------------------------------
__global__ __launch_bounds__(BLOCK_THREADS)
void egnn_edge_kernel(const float* __restrict__ h, const float* __restrict__ x,
                      const float* __restrict__ edge_attr,
                      const int* __restrict__ eidx,
                      const __bf16* We0, const __bf16* We1, const __bf16* We2,
                      const __bf16* Wa0, const __bf16* Wa1,
                      const __bf16* Wc0, const __bf16* Wc1, const __bf16* Wc2,
                      const float* be0, const float* be1, const float* be2,
                      const float* ba0, const float* ba1,
                      const float* bc0, const float* bc1, const float* bc2,
                      float* __restrict__ node_aggr, float* __restrict__ x_out) {
  __shared__ __bf16 s_buf0[WAVES_PER_BLOCK][16 * EDGE_ASTRIDE];
  __shared__ __bf16 s_buf1[WAVES_PER_BLOCK][16 * EDGE_ASTRIDE];
  __shared__ int    s_row [WAVES_PER_BLOCK][16];
  __shared__ int    s_col [WAVES_PER_BLOCK][16];
  __shared__ float  s_relx[WAVES_PER_BLOCK][16][COORDS];
  __shared__ float  s_dist[WAVES_PER_BLOCK][16];
  __shared__ float  s_attn[WAVES_PER_BLOCK][16];

  const int lane = threadIdx.x & 31;
  const int w    = threadIdx.x >> 5;
  const int e0   = (blockIdx.x * WAVES_PER_BLOCK + w) * 16;

  __bf16* buf0 = s_buf0[w];
  __bf16* buf1 = s_buf1[w];

  // per-edge metadata (lanes 0..15 each own one edge)
  if (lane < 16) {
    const int e  = e0 + lane;
    const int ri = eidx[e];
    const int ci = eidx[N_EDGES + e];
    s_row[w][lane] = ri;
    s_col[w][lane] = ci;
    const float rx = x[ri * COORDS + 0] - x[ci * COORDS + 0];
    const float ry = x[ri * COORDS + 1] - x[ci * COORDS + 1];
    s_relx[w][lane][0] = rx;
    s_relx[w][lane][1] = ry;
    s_dist[w][lane] = sqrtf(rx * rx + ry * ry);
  }
  lds_fence();

  // gather input tile [16 x 160] bf16: edge_attr(16) | h[row](64) | h[col](64) | dist | pad
  {
    const int r    = lane >> 1;
    const int half = lane & 1;
    const int e    = e0 + r;
    __bf16* dst    = buf0 + r * EDGE_ASTRIDE;
    if (half == 0) {
      const float4* ea = (const float4*)(edge_attr + (size_t)e * EDGE_DIM);
#pragma unroll
      for (int q = 0; q < EDGE_DIM / 4; ++q) {
        float4 v = ea[q];
        __bf16* p = dst + q * 4;
        p[0] = (__bf16)v.x; p[1] = (__bf16)v.y; p[2] = (__bf16)v.z; p[3] = (__bf16)v.w;
      }
      const float4* hr = (const float4*)(h + (size_t)s_row[w][r] * NODE_DIM);
#pragma unroll
      for (int q = 0; q < NODE_DIM / 4; ++q) {
        float4 v = hr[q];
        __bf16* p = dst + EDGE_DIM + q * 4;
        p[0] = (__bf16)v.x; p[1] = (__bf16)v.y; p[2] = (__bf16)v.z; p[3] = (__bf16)v.w;
      }
    } else {
      const float4* hc = (const float4*)(h + (size_t)s_col[w][r] * NODE_DIM);
#pragma unroll
      for (int q = 0; q < NODE_DIM / 4; ++q) {
        float4 v = hc[q];
        __bf16* p = dst + EDGE_DIM + NODE_DIM + q * 4;
        p[0] = (__bf16)v.x; p[1] = (__bf16)v.y; p[2] = (__bf16)v.z; p[3] = (__bf16)v.w;
      }
      dst[144] = (__bf16)s_dist[w][r];
#pragma unroll
      for (int c = 145; c < 160; ++c) dst[c] = (__bf16)0.f;
    }
  }
  lds_fence();

  // edge MLP: 160->128 silu, 128->128 silu, 128->128 (+bias, no act)
  mlp_layer<5, 8, true >(buf0, EDGE_ASTRIDE, buf1, EDGE_ASTRIDE, We0, be0, lane);
  mlp_layer<4, 8, true >(buf1, EDGE_ASTRIDE, buf0, EDGE_ASTRIDE, We1, be1, lane);
  mlp_layer<4, 8, false>(buf0, EDGE_ASTRIDE, buf1, EDGE_ASTRIDE, We2, be2, lane);
  // buf1 = edge_out (bf16)

  // attention gate: sigmoid( silu(edge_out@Wa0 + ba0) @ Wa1 + ba1 )
  mlp_layer<4, 8, true>(buf1, EDGE_ASTRIDE, buf0, EDGE_ASTRIDE, Wa0, ba0, lane);
  {
    v8f acc[1];
    wmma_gemm<4, 1>(buf0, EDGE_ASTRIDE, Wa1, lane, acc);   // N padded to 16, col 0 valid
    if ((lane & 15) == 0) {
      const float ba = ba1[0];
      const int mb = (lane >> 4) << 3;
#pragma unroll
      for (int r = 0; r < 8; ++r) {
        const float logit = acc[0][r] + ba;
        s_attn[w][mb + r] = 1.f / (1.f + __expf(-logit));
      }
    }
    lds_fence();
  }

  // scale edge_out by attn; atomically aggregate into node_aggr (L2-resident);
  // stage scaled bf16 into buf0 as coord-MLP input
  {
    const int nlo = lane & 15;
    const int mb  = (lane >> 4) << 3;
#pragma unroll
    for (int r = 0; r < 8; ++r) {
      const int   m   = mb + r;
      const float a   = s_attn[w][m];
      const int   dst = s_row[w][m] * HIDDEN;
#pragma unroll
      for (int nt = 0; nt < 8; ++nt) {
        const int   n = (nt << 4) + nlo;
        const float v = (float)buf1[m * EDGE_ASTRIDE + n] * a;
        __hip_atomic_fetch_add(&node_aggr[dst + n], v,
                               __ATOMIC_RELAXED, __HIP_MEMORY_SCOPE_AGENT);
        buf0[m * EDGE_ASTRIDE + n] = (__bf16)v;
      }
    }
  }
  lds_fence();

  // coord MLP: 128->128 silu, 128->128 silu, 128->2 (N padded to 16)
  mlp_layer<4, 8, true>(buf0, EDGE_ASTRIDE, buf1, EDGE_ASTRIDE, Wc0, bc0, lane);
  mlp_layer<4, 8, true>(buf1, EDGE_ASTRIDE, buf0, EDGE_ASTRIDE, Wc1, bc1, lane);
  {
    v8f acc[1];
    wmma_gemm<4, 1>(buf0, EDGE_ASTRIDE, Wc2, lane, acc);
    const int nlo = lane & 15;
    if (nlo < COORDS) {
      const float bc = bc2[nlo];
      const int mb = (lane >> 4) << 3;
#pragma unroll
      for (int r = 0; r < 8; ++r) {
        const int   m   = mb + r;
        const float v   = acc[0][r] + bc;
        const float msg = v * s_relx[w][m][nlo] / (s_dist[w][m] + 1e-8f);
        __hip_atomic_fetch_add(&x_out[s_row[w][m] * COORDS + nlo], msg,
                               __ATOMIC_RELAXED, __HIP_MEMORY_SCOPE_AGENT);
      }
    }
  }
}

// ---------------------------------------------------------------------------
// Node kernel: one wave = 16 nodes. h_out = h + MLP([h | node_aggr])
// ---------------------------------------------------------------------------
__global__ __launch_bounds__(BLOCK_THREADS)
void egnn_node_kernel(const float* __restrict__ h,
                      const float* __restrict__ node_aggr,
                      const __bf16* Wn0, const __bf16* Wn1, const __bf16* Wn2,
                      const float* bn0, const float* bn1, const float* bn2,
                      float* __restrict__ h_out) {
  __shared__ __bf16 s_buf0[WAVES_PER_BLOCK][16 * NODE_ASTRIDE];
  __shared__ __bf16 s_buf1[WAVES_PER_BLOCK][16 * EDGE_ASTRIDE];

  const int lane   = threadIdx.x & 31;
  const int w      = threadIdx.x >> 5;
  const int tile   = blockIdx.x * WAVES_PER_BLOCK + w;
  const int ntiles = (N_NODES + 15) / 16;
  if (tile >= ntiles) return;          // wave-uniform guard (no WMMA below with partial EXEC)
  const int n0 = tile * 16;

  __bf16* buf0 = s_buf0[w];
  __bf16* buf1 = s_buf1[w];

  // build input [16 x 192]: h(64) | node_aggr(128)
  {
    const int r    = lane >> 1;
    const int half = lane & 1;
    const int node = n0 + r;
    __bf16* dst = buf0 + r * NODE_ASTRIDE;
    const float4* ag = (const float4*)(node_aggr + (size_t)node * HIDDEN);
    if (half == 0) {
      const float4* hr = (const float4*)(h + (size_t)node * NODE_DIM);
#pragma unroll
      for (int q = 0; q < NODE_DIM / 4; ++q) {
        float4 v = hr[q];
        __bf16* p = dst + q * 4;
        p[0] = (__bf16)v.x; p[1] = (__bf16)v.y; p[2] = (__bf16)v.z; p[3] = (__bf16)v.w;
      }
#pragma unroll
      for (int q = 0; q < 8; ++q) {          // aggr cols 0..31
        float4 v = ag[q];
        __bf16* p = dst + NODE_DIM + q * 4;
        p[0] = (__bf16)v.x; p[1] = (__bf16)v.y; p[2] = (__bf16)v.z; p[3] = (__bf16)v.w;
      }
    } else {
#pragma unroll
      for (int q = 8; q < 32; ++q) {         // aggr cols 32..127
        float4 v = ag[q];
        __bf16* p = dst + NODE_DIM + q * 4;
        p[0] = (__bf16)v.x; p[1] = (__bf16)v.y; p[2] = (__bf16)v.z; p[3] = (__bf16)v.w;
      }
    }
  }
  lds_fence();

  mlp_layer<6, 8, true>(buf0, NODE_ASTRIDE, buf1, EDGE_ASTRIDE, Wn0, bn0, lane);
  mlp_layer<4, 8, true>(buf1, EDGE_ASTRIDE, buf0, NODE_ASTRIDE, Wn1, bn1, lane);
  {
    v8f acc[4];
    wmma_gemm<4, 4>(buf0, NODE_ASTRIDE, Wn2, lane, acc);   // 128 -> 64
    const int nlo = lane & 15;
    const int mb  = (lane >> 4) << 3;
#pragma unroll
    for (int nt = 0; nt < 4; ++nt) {
      const int   n  = (nt << 4) + nlo;
      const float bv = bn2[n];
#pragma unroll
      for (int r = 0; r < 8; ++r) {
        const int node = n0 + mb + r;
        h_out[(size_t)node * NODE_DIM + n] =
            h[(size_t)node * NODE_DIM + n] + acc[nt][r] + bv;
      }
    }
  }
}

// ---------------------------------------------------------------------------
// Prep: swizzle fp32 [K,N] weights into per-fragment bf16 B-layout
// (frag f = kt*NT+nt; per lane 16 contiguous bf16: k = kt*32 + (e>=8)*16 + (lane>=16)*8 + (e&7),
//  n = nt*16 + (lane&15)); zero-pad outside [K,N].
// ---------------------------------------------------------------------------
__global__ void swizzle_w_kernel(const float* __restrict__ wf, __bf16* __restrict__ out,
                                 int K, int N, int KT, int NT) {
  const int idx   = blockIdx.x * 256 + threadIdx.x;
  const int total = KT * NT * 32 * 16;
  if (idx >= total) return;
  const int e    = idx & 15;
  const int lane = (idx >> 4) & 31;
  const int f    = idx >> 9;
  const int nt   = f % NT;
  const int kt   = f / NT;
  const int n = nt * 16 + (lane & 15);
  const int k = kt * 32 + ((e >> 3) << 4) + ((lane >> 4) << 3) + (e & 7);
  const float v = (k < K && n < N) ? wf[k * N + n] : 0.f;
  out[idx] = (__bf16)v;
}

__global__ void init_kernel(float* __restrict__ node_aggr,
                            float* __restrict__ x_out,
                            const float* __restrict__ x) {
  const int i = blockIdx.x * 256 + threadIdx.x;
  if (i < (N_NODES * HIDDEN) / 4)
    ((float4*)node_aggr)[i] = make_float4(0.f, 0.f, 0.f, 0.f);
  if (i < N_NODES * COORDS) x_out[i] = x[i];   // x_out starts as x, atomics add coord_aggr
}

// ---------------------------------------------------------------------------
extern "C" void kernel_launch(void* const* d_in, const int* in_sizes, int n_in,
                              void* d_out, int out_size, void* d_ws, size_t ws_size,
                              hipStream_t stream) {
  (void)in_sizes; (void)n_in; (void)out_size; (void)ws_size;

  const float* h         = (const float*)d_in[0];
  const float* x         = (const float*)d_in[1];
  const float* edge_attr = (const float*)d_in[2];
  const int*   eidx      = (const int*)d_in[3];
  // params in dict insertion order: edge, node, coord, attn; per layer: w, b
  const float* We0f = (const float*)d_in[4];  const float* be0 = (const float*)d_in[5];
  const float* We1f = (const float*)d_in[6];  const float* be1 = (const float*)d_in[7];
  const float* We2f = (const float*)d_in[8];  const float* be2 = (const float*)d_in[9];
  const float* Wn0f = (const float*)d_in[10]; const float* bn0 = (const float*)d_in[11];
  const float* Wn1f = (const float*)d_in[12]; const float* bn1 = (const float*)d_in[13];
  const float* Wn2f = (const float*)d_in[14]; const float* bn2 = (const float*)d_in[15];
  const float* Wc0f = (const float*)d_in[16]; const float* bc0 = (const float*)d_in[17];
  const float* Wc1f = (const float*)d_in[18]; const float* bc1 = (const float*)d_in[19];
  const float* Wc2f = (const float*)d_in[20]; const float* bc2 = (const float*)d_in[21];
  const float* Wa0f = (const float*)d_in[22]; const float* ba0 = (const float*)d_in[23];
  const float* Wa1f = (const float*)d_in[24]; const float* ba1 = (const float*)d_in[25];

  float* h_out = (float*)d_out;
  float* x_out = (float*)d_out + (size_t)N_NODES * NODE_DIM;

  // workspace: [node_aggr fp32 25.6MB][swizzled bf16 weights ~304KB]
  char*  ws = (char*)d_ws;
  float* node_aggr = (float*)ws;
  size_t off = (size_t)N_NODES * HIDDEN * sizeof(float);
  auto alloc_bf16 = [&](size_t elems) {
    __bf16* p = (__bf16*)(ws + off);
    off += elems * sizeof(__bf16);
    return p;
  };
  __bf16* We0 = alloc_bf16(5 * 8 * 512);
  __bf16* We1 = alloc_bf16(4 * 8 * 512);
  __bf16* We2 = alloc_bf16(4 * 8 * 512);
  __bf16* Wn0 = alloc_bf16(6 * 8 * 512);
  __bf16* Wn1 = alloc_bf16(4 * 8 * 512);
  __bf16* Wn2 = alloc_bf16(4 * 4 * 512);
  __bf16* Wc0 = alloc_bf16(4 * 8 * 512);
  __bf16* Wc1 = alloc_bf16(4 * 8 * 512);
  __bf16* Wc2 = alloc_bf16(4 * 1 * 512);
  __bf16* Wa0 = alloc_bf16(4 * 8 * 512);
  __bf16* Wa1 = alloc_bf16(4 * 1 * 512);

  auto swz = [&](const float* wf, __bf16* wo, int K, int N, int KT, int NT) {
    const int total = KT * NT * 512;
    swizzle_w_kernel<<<(total + 255) / 256, 256, 0, stream>>>(wf, wo, K, N, KT, NT);
  };
  swz(We0f, We0, 145, 128, 5, 8);
  swz(We1f, We1, 128, 128, 4, 8);
  swz(We2f, We2, 128, 128, 4, 8);
  swz(Wn0f, Wn0, 192, 128, 6, 8);
  swz(Wn1f, Wn1, 128, 128, 4, 8);
  swz(Wn2f, Wn2, 128,  64, 4, 4);
  swz(Wc0f, Wc0, 128, 128, 4, 8);
  swz(Wc1f, Wc1, 128, 128, 4, 8);
  swz(Wc2f, Wc2, 128,   2, 4, 1);
  swz(Wa0f, Wa0, 128, 128, 4, 8);
  swz(Wa1f, Wa1, 128,   1, 4, 1);

  init_kernel<<<((N_NODES * HIDDEN / 4) + 255) / 256, 256, 0, stream>>>(
      node_aggr, x_out, x);

  const int edge_blocks = N_EDGES / (16 * WAVES_PER_BLOCK);   // 800000/64 = 12500
  egnn_edge_kernel<<<edge_blocks, BLOCK_THREADS, 0, stream>>>(
      h, x, edge_attr, eidx,
      We0, We1, We2, Wa0, Wa1, Wc0, Wc1, Wc2,
      be0, be1, be2, ba0, ba1, bc0, bc1, bc2,
      node_aggr, x_out);

  const int node_tiles  = (N_NODES + 15) / 16;                // 3125
  const int node_blocks = (node_tiles + WAVES_PER_BLOCK - 1) / WAVES_PER_BLOCK;
  egnn_node_kernel<<<node_blocks, BLOCK_THREADS, 0, stream>>>(
      h, node_aggr, Wn0, Wn1, Wn2, bn0, bn1, bn2, h_out);
}
